// TemporalGATGRU_15264313770442
// MI455X (gfx1250) — matrix operand order repaired and checked by reading, not compiled, not run on hardware
//
#include <hip/hip_runtime.h>
#include <stdint.h>

// ---------------- problem constants ----------------
#define N_NODES   50000
#define N_EDGES   800000
#define E_TOT     (N_EDGES + N_NODES)   // edges + self loops
#define F_IN      128
#define HID       128
#define HEADS     4
#define HC1       (HEADS * HID)         // 512
#define N_GRAPHS  64
#define NEG_SLOPE 0.2f

// ---------------- WMMA vector types ----------------
typedef __attribute__((ext_vector_type(16))) __bf16       v16bf;
typedef __attribute__((ext_vector_type(8)))  float        v8f;
typedef __attribute__((ext_vector_type(4)))  unsigned int v4u;
typedef __attribute__((ext_vector_type(2)))  unsigned int v2u;

union FragU { v4u u[2]; v16bf bf; };

// ---------------- bf16 helpers (bit-exact, storage = ushort) ----------------
__device__ __forceinline__ float bf2f(unsigned short h) {
  return __uint_as_float(((unsigned int)h) << 16);
}
__device__ __forceinline__ unsigned short f2bf(float f) {
  unsigned int u = __float_as_uint(f);
  u += 0x7FFFu + ((u >> 16) & 1u);   // round-to-nearest-even
  return (unsigned short)(u >> 16);
}

// monotone float<->uint encoding for atomicMax on floats
__device__ __forceinline__ unsigned int fenc(float x) {
  unsigned int u = __float_as_uint(x);
  return (u & 0x80000000u) ? ~u : (u | 0x80000000u);
}
__device__ __forceinline__ float fdec(unsigned int u) {
  return __uint_as_float((u & 0x80000000u) ? (u ^ 0x80000000u) : ~u);
}

// ---------------- conversion kernels ----------------
__global__ void k_cvt_bf16(const float* __restrict__ in,
                           unsigned short* __restrict__ out, int n) {
  int i = blockIdx.x * blockDim.x + threadIdx.x;
  if (i < n) out[i] = f2bf(in[i]);
}

// W [K,N] f32 row-major  ->  out [N,K] bf16 row-major (B^T for WMMA)
__global__ void k_transpose_bf16(const float* __restrict__ W,
                                 unsigned short* __restrict__ out, int K, int N) {
  int i = blockIdx.x * blockDim.x + threadIdx.x;
  if (i < K * N) {
    int k = i / N, n = i - k * N;
    out[(size_t)n * K + k] = f2bf(W[i]);
  }
}

// ---------------- WMMA GEMM: C[M,N] = A[M,K] * B[K,N], B given transposed --------
// Block = 8 waves computing a 16x128 output slab. The shared 16xK A tile is staged
// into LDS ONCE per block via CDNA5 async-to-LDS DMA (ASYNCcnt), then each wave
// builds its WMMA fragments with ds_load_b128. B comes straight from global (tiny,
// L2-resident). M = 3125*16 exactly -> no bounds checks, EXEC all-ones for WMMA.
template <int KIT>  // K = KIT*32
__global__ __launch_bounds__(256) void k_gemm_bf16(
    const unsigned short* __restrict__ A,   // [M, K] bf16
    const unsigned short* __restrict__ Bt,  // [N, K] bf16 (transposed B)
    unsigned short* __restrict__ C,         // [M, N] bf16
    int Ncols) {
  constexpr int K      = KIT * 32;
  constexpr int ROWB   = K * 2;          // bytes per A row
  constexpr int STRIDE = ROWB + 16;      // padded LDS row stride (bank stagger)
  constexpr int CHROW  = ROWB / 16;      // 16B chunks per row
  constexpr int CHTOT  = 16 * CHROW;
  __shared__ __align__(16) unsigned char sA[16 * STRIDE];

  const int tid  = threadIdx.x;
  const int lane = tid & 31;
  const int wave = tid >> 5;
  const int half = lane >> 4;
  const int r    = lane & 15;
  const int tm   = blockIdx.x;              // 16-row tile index
  const int tn   = blockIdx.y * 8 + wave;   // 16-col tile index

  // async-stage the 16xK A tile into LDS: one 16B chunk per active lane
  const unsigned char* abase = (const unsigned char*)(A + (size_t)tm * 16 * K);
  for (int i = tid; i < CHTOT; i += 256) {
    const int rr = i / CHROW, ss = i - rr * CHROW;
    unsigned ldsoff = (unsigned)(uintptr_t)(&sA[rr * STRIDE + ss * 16]);
    const unsigned char* ga = abase + rr * ROWB + ss * 16;
    asm volatile("global_load_async_to_lds_b128 %0, %1, off"
                 :: "v"(ldsoff), "v"(ga) : "memory");
  }
  asm volatile("s_wait_asynccnt 0" ::: "memory");
  __syncthreads();

  const unsigned char*  arow = &sA[r * STRIDE];
  const unsigned short* brow = Bt + (size_t)(tn * 16 + r) * K;

  v8f acc = {0.f, 0.f, 0.f, 0.f, 0.f, 0.f, 0.f, 0.f};
#pragma unroll
  for (int kk = 0; kk < KIT; ++kk) {
    const int k0 = kk * 32;
    FragU a, b;
    // A lane layout: rows striped, K in {half*8..+7} u {16+half*8..+7}
    a.u[0] = *(const v4u*)(arow + (k0 + half * 8) * 2);
    a.u[1] = *(const v4u*)(arow + (k0 + 16 + half * 8) * 2);
    // B lane layout: cols striped, K in {half*8..+7} u {16+half*8..+7}
    b.u[0] = *(const v4u*)(brow + k0 + half * 8);
    b.u[1] = *(const v4u*)(brow + k0 + 16 + half * 8);
    acc = __builtin_amdgcn_wmma_f32_16x16x32_bf16(
        false, a.bf, false, b.bf, (short)0, acc, false, false);
  }
  // D layout: VGPR j -> row j + 8*half, col = lane&15
  unsigned short* crow = C + (size_t)(tm * 16 + 8 * half) * Ncols + tn * 16 + r;
#pragma unroll
  for (int j = 0; j < 8; ++j) crow[(size_t)j * Ncols] = f2bf(acc[j]);
}

// ---------------- attention scores: a_s[n,h]=h[n,h,:].att_s[h], same for dst ------
template <int H>
__global__ void k_attn_scores(const unsigned short* __restrict__ Hfeat, // [N, H*128]
                              const float* __restrict__ att_s,          // [H,128]
                              const float* __restrict__ att_d,
                              float* __restrict__ as_, float* __restrict__ ad_) {
  int wave = threadIdx.x >> 5, lane = threadIdx.x & 31;
  int node = blockIdx.x * (blockDim.x >> 5) + wave;
  if (node >= N_NODES) return;
  const unsigned short* hrow = Hfeat + (size_t)node * (H * 128);
#pragma unroll
  for (int h = 0; h < H; ++h) {
    float ss = 0.f, sd = 0.f;
#pragma unroll
    for (int j = 0; j < 4; ++j) {
      int c = h * 128 + j * 32 + lane;
      float v = bf2f(hrow[c]);
      ss += v * att_s[h * 128 + j * 32 + lane];
      sd += v * att_d[h * 128 + j * 32 + lane];
    }
#pragma unroll
    for (int m = 16; m > 0; m >>= 1) {
      ss += __shfl_xor(ss, m);
      sd += __shfl_xor(sd, m);
    }
    if (lane == 0) { as_[node * H + h] = ss; ad_[node * H + h] = sd; }
  }
}

// ---------------- edge pass 1: segment max per (dst, head) ----------------
template <int H>
__global__ void k_edge_max(const int* __restrict__ ei,
                           const float* __restrict__ as_, const float* __restrict__ ad_,
                           unsigned int* __restrict__ m) {
  int e = blockIdx.x * blockDim.x + threadIdx.x;
  if (e >= E_TOT) return;
  int s, d;
  if (e < N_EDGES) { s = ei[e]; d = ei[N_EDGES + e]; }
  else             { s = d = e - N_EDGES; }
#pragma unroll
  for (int h = 0; h < H; ++h) {
    float a = as_[s * H + h] + ad_[d * H + h];
    a = a > 0.f ? a : NEG_SLOPE * a;
    atomicMax(&m[d * H + h], fenc(a));
  }
}

// ---------------- edge pass 2 (fused): denom += e ; agg[d] += e * h[s] -----------
// one wave per edge; each lane owns a CONTIGUOUS feature chunk (vectorized b128/b64
// gathers); softmax normalization deferred to the per-node finalize pass
template <int H>
__global__ void k_edge_msg(const int* __restrict__ ei,
                           const float* __restrict__ as_, const float* __restrict__ ad_,
                           const unsigned int* __restrict__ m,
                           float* __restrict__ denom,
                           const unsigned short* __restrict__ Hfeat, // [N, H*128] bf16
                           float* __restrict__ agg) {                // [N, H*128] f32
  int wave = threadIdx.x >> 5, lane = threadIdx.x & 31;
  long e = (long)blockIdx.x * (blockDim.x >> 5) + wave;
  if (e >= E_TOT) return;
  int s, d;
  if (e < N_EDGES) { s = ei[e]; d = ei[N_EDGES + e]; }
  else             { s = d = (int)(e - N_EDGES); }
  const unsigned short* hrow = Hfeat + (size_t)s * (H * 128);
  float* arow = agg + (size_t)d * (H * 128);

  // per-head denominator contribution (lanes 0..H-1)
  if (lane < H) {
    float a = as_[s * H + lane] + ad_[d * H + lane];
    a = a > 0.f ? a : NEG_SLOPE * a;
    atomicAdd(&denom[d * H + lane], __expf(a - fdec(m[d * H + lane])));
  }

  // this lane's head (constant over its contiguous chunk)
  const int head = (H == 4) ? (lane >> 3) : 0;
  float a = as_[s * H + head] + ad_[d * H + head];
  a = a > 0.f ? a : NEG_SLOPE * a;
  const float ex = __expf(a - fdec(m[d * H + head]));

  if constexpr (H == 4) {
    // 512 cols / 32 lanes = 16 bf16 per lane = two b128 gathers
    const v4u* p = (const v4u*)(hrow + lane * 16);
    v4u u0 = p[0], u1 = p[1];
    float* dst = arow + lane * 16;
#pragma unroll
    for (int t = 0; t < 4; ++t) {
      unsigned u = u0[t];
      atomicAdd(&dst[2 * t],     ex * bf2f((unsigned short)(u & 0xFFFFu)));
      atomicAdd(&dst[2 * t + 1], ex * bf2f((unsigned short)(u >> 16)));
    }
#pragma unroll
    for (int t = 0; t < 4; ++t) {
      unsigned u = u1[t];
      atomicAdd(&dst[8 + 2 * t],     ex * bf2f((unsigned short)(u & 0xFFFFu)));
      atomicAdd(&dst[8 + 2 * t + 1], ex * bf2f((unsigned short)(u >> 16)));
    }
  } else {
    // 128 cols / 32 lanes = 4 bf16 per lane = one b64 gather
    v2u u = *(const v2u*)(hrow + lane * 4);
    float* dst = arow + lane * 4;
#pragma unroll
    for (int t = 0; t < 2; ++t) {
      unsigned uu = u[t];
      atomicAdd(&dst[2 * t],     ex * bf2f((unsigned short)(uu & 0xFFFFu)));
      atomicAdd(&dst[2 * t + 1], ex * bf2f((unsigned short)(uu >> 16)));
    }
  }
}

// ---------------- finalize layer1: /denom, +bias, relu, ->bf16 ----------------
__global__ void k_finalize1(const float* __restrict__ agg, const float* __restrict__ denom,
                            const float* __restrict__ b1, unsigned short* __restrict__ out) {
  long i = (long)blockIdx.x * blockDim.x + threadIdx.x;
  if (i >= (long)N_NODES * HC1) return;
  int n = (int)(i >> 9);        // /512
  int c = (int)(i & 511);
  int h = c >> 7;
  float v = agg[i] / denom[n * HEADS + h] + b1[c];
  out[i] = f2bf(v > 0.f ? v : 0.f);
}

// ---------------- finalize layer2 + global mean-pool numerator ----------------
__global__ void k_finalize2_pool(const float* __restrict__ agg, const float* __restrict__ denom,
                                 const float* __restrict__ b2, const int* __restrict__ batch,
                                 float* __restrict__ pool) {
  long i = (long)blockIdx.x * blockDim.x + threadIdx.x;
  if (i >= (long)N_NODES * HID) return;
  int n = (int)(i >> 7);
  int c = (int)(i & 127);
  float v = agg[i] / denom[n] + b2[c];
  atomicAdd(&pool[(size_t)batch[n] * HID + c], v);
}

__global__ void k_counts(const int* __restrict__ batch, float* __restrict__ cnt) {
  int n = blockIdx.x * blockDim.x + threadIdx.x;
  if (n < N_NODES) atomicAdd(&cnt[batch[n]], 1.0f);
}

// ---------------- GRU (h0 = 0 -> gh = bhh exactly) + FC head ----------------
__global__ __launch_bounds__(128) void k_gru_fc(
    const float* __restrict__ pool, const float* __restrict__ cnt,
    const float* __restrict__ Wih0, const float* __restrict__ bih0, const float* __restrict__ bhh0,
    const float* __restrict__ Wih1, const float* __restrict__ bih1, const float* __restrict__ bhh1,
    const float* __restrict__ Wfc,  const float* __restrict__ bfc,  float* __restrict__ out) {
  __shared__ float xin[HID], xmid[HID];
  int g = blockIdx.x, j = threadIdx.x;
  float c = cnt[g]; c = c > 1.f ? c : 1.f;
  xin[j] = pool[(size_t)g * HID + j] / c;
  __syncthreads();
  // layer 0
  float ir = bih0[j], iz = bih0[HID + j], in_ = bih0[2 * HID + j];
  for (int k = 0; k < HID; ++k) {
    float x = xin[k];
    ir  += x * Wih0[(size_t)j * HID + k];
    iz  += x * Wih0[(size_t)(HID + j) * HID + k];
    in_ += x * Wih0[(size_t)(2 * HID + j) * HID + k];
  }
  float r  = 1.f / (1.f + __expf(-(ir + bhh0[j])));
  float z  = 1.f / (1.f + __expf(-(iz + bhh0[HID + j])));
  float nn = tanhf(in_ + r * bhh0[2 * HID + j]);
  float hv = (1.f - z) * nn;      // + z*h0, h0 = 0
  xmid[j] = hv;
  __syncthreads();
  // layer 1
  ir = bih1[j]; iz = bih1[HID + j]; in_ = bih1[2 * HID + j];
  for (int k = 0; k < HID; ++k) {
    float x = xmid[k];
    ir  += x * Wih1[(size_t)j * HID + k];
    iz  += x * Wih1[(size_t)(HID + j) * HID + k];
    in_ += x * Wih1[(size_t)(2 * HID + j) * HID + k];
  }
  r  = 1.f / (1.f + __expf(-(ir + bhh1[j])));
  z  = 1.f / (1.f + __expf(-(iz + bhh1[HID + j])));
  nn = tanhf(in_ + r * bhh1[2 * HID + j]);
  hv = (1.f - z) * nn;
  __syncthreads();
  xin[j] = hv;
  __syncthreads();
  if (j < 3) {
    float o = bfc[j];
    for (int k = 0; k < HID; ++k) o += xin[k] * Wfc[(size_t)j * HID + k];
    out[g * 3 + j] = o;
  }
}

// ---------------- workspace layout ----------------
constexpr size_t al256(size_t x) { return (x + 255) & ~(size_t)255; }
constexpr size_t OFF_XB    = 0;                                                  // x bf16
constexpr size_t OFF_W1T   = OFF_XB    + al256((size_t)N_NODES * F_IN * 2);
constexpr size_t OFF_W2T   = OFF_W1T   + al256((size_t)HC1 * F_IN * 2);
constexpr size_t OFF_H1B   = OFF_W2T   + al256((size_t)HID * HC1 * 2);           // layer1 h bf16
constexpr size_t OFF_A1S   = OFF_H1B   + al256((size_t)N_NODES * HC1 * 2);
constexpr size_t OFF_A1D   = OFF_A1S   + al256((size_t)N_NODES * HEADS * 4);
constexpr size_t OFF_M1    = OFF_A1D   + al256((size_t)N_NODES * HEADS * 4);
constexpr size_t OFF_D1    = OFF_M1    + al256((size_t)N_NODES * HEADS * 4);
constexpr size_t OFF_AGG1  = OFF_D1    + al256((size_t)N_NODES * HEADS * 4);
constexpr size_t OFF_RELU1 = OFF_AGG1  + al256((size_t)N_NODES * HC1 * 4);       // relu out bf16
constexpr size_t OFF_H2B   = OFF_RELU1 + al256((size_t)N_NODES * HC1 * 2);       // layer2 h bf16
constexpr size_t OFF_A2S   = OFF_H2B   + al256((size_t)N_NODES * HID * 2);
constexpr size_t OFF_A2D   = OFF_A2S   + al256((size_t)N_NODES * 4);
constexpr size_t OFF_M2    = OFF_A2D   + al256((size_t)N_NODES * 4);
constexpr size_t OFF_D2    = OFF_M2    + al256((size_t)N_NODES * 4);
constexpr size_t OFF_AGG2  = OFF_D2    + al256((size_t)N_NODES * 4);
constexpr size_t OFF_POOL  = OFF_AGG2  + al256((size_t)N_NODES * HID * 4);
constexpr size_t OFF_CNT   = OFF_POOL  + al256((size_t)N_GRAPHS * HID * 4);

extern "C" void kernel_launch(void* const* d_in, const int* in_sizes, int n_in,
                              void* d_out, int out_size, void* d_ws, size_t ws_size,
                              hipStream_t stream) {
  (void)in_sizes; (void)n_in; (void)out_size; (void)ws_size;
  const float* x        = (const float*)d_in[0];
  const int*   ei       = (const int*)  d_in[1];
  const int*   batch    = (const int*)  d_in[2];
  const float* W1       = (const float*)d_in[3];
  const float* att_src1 = (const float*)d_in[4];
  const float* att_dst1 = (const float*)d_in[5];
  const float* b1       = (const float*)d_in[6];
  const float* W2       = (const float*)d_in[7];
  const float* att_src2 = (const float*)d_in[8];
  const float* att_dst2 = (const float*)d_in[9];
  const float* b2       = (const float*)d_in[10];
  const float* Wih0     = (const float*)d_in[11];
  const float* bih0     = (const float*)d_in[13];
  const float* bhh0     = (const float*)d_in[14];
  const float* Wih1     = (const float*)d_in[15];
  const float* bih1     = (const float*)d_in[17];
  const float* bhh1     = (const float*)d_in[18];
  const float* Wfc      = (const float*)d_in[19];
  const float* bfc      = (const float*)d_in[20];
  float* out = (float*)d_out;

  char* ws = (char*)d_ws;
  unsigned short* xb    = (unsigned short*)(ws + OFF_XB);
  unsigned short* w1t   = (unsigned short*)(ws + OFF_W1T);
  unsigned short* w2t   = (unsigned short*)(ws + OFF_W2T);
  unsigned short* h1b   = (unsigned short*)(ws + OFF_H1B);
  float*          a1s   = (float*)(ws + OFF_A1S);
  float*          a1d   = (float*)(ws + OFF_A1D);
  unsigned int*   m1    = (unsigned int*)(ws + OFF_M1);
  float*          d1    = (float*)(ws + OFF_D1);
  float*          agg1  = (float*)(ws + OFF_AGG1);
  unsigned short* relu1 = (unsigned short*)(ws + OFF_RELU1);
  unsigned short* h2b   = (unsigned short*)(ws + OFF_H2B);
  float*          a2s   = (float*)(ws + OFF_A2S);
  float*          a2d   = (float*)(ws + OFF_A2D);
  unsigned int*   m2    = (unsigned int*)(ws + OFF_M2);
  float*          d2    = (float*)(ws + OFF_D2);
  float*          agg2  = (float*)(ws + OFF_AGG2);
  float*          pool  = (float*)(ws + OFF_POOL);
  float*          cnt   = (float*)(ws + OFF_CNT);

  // deterministic zero-init of accumulators (capture-safe)
  hipMemsetAsync(m1,   0, (size_t)N_NODES * HEADS * 4, stream);
  hipMemsetAsync(d1,   0, (size_t)N_NODES * HEADS * 4, stream);
  hipMemsetAsync(agg1, 0, (size_t)N_NODES * HC1 * 4,   stream);
  hipMemsetAsync(m2,   0, (size_t)N_NODES * 4,         stream);
  hipMemsetAsync(d2,   0, (size_t)N_NODES * 4,         stream);
  hipMemsetAsync(agg2, 0, (size_t)N_NODES * HID * 4,   stream);
  hipMemsetAsync(pool, 0, (size_t)N_GRAPHS * HID * 4,  stream);
  hipMemsetAsync(cnt,  0, (size_t)N_GRAPHS * 4,        stream);

  // conversions
  {
    int n = N_NODES * F_IN;
    k_cvt_bf16<<<(n + 255) / 256, 256, 0, stream>>>(x, xb, n);
  }
  k_transpose_bf16<<<(F_IN * HC1 + 255) / 256, 256, 0, stream>>>(W1, w1t, F_IN, HC1);
  k_transpose_bf16<<<(HC1 * HID + 255) / 256, 256, 0, stream>>>(W2, w2t, HC1, HID);

  // ---- GAT layer 1 ----
  k_gemm_bf16<4><<<dim3(N_NODES / 16, HC1 / 128), 256, 0, stream>>>(xb, w1t, h1b, HC1);
  k_attn_scores<HEADS><<<(N_NODES + 7) / 8, 256, 0, stream>>>(h1b, att_src1, att_dst1, a1s, a1d);
  k_edge_max<HEADS><<<(E_TOT + 255) / 256, 256, 0, stream>>>(ei, a1s, a1d, m1);
  k_edge_msg<HEADS><<<(E_TOT + 7) / 8, 256, 0, stream>>>(ei, a1s, a1d, m1, d1, h1b, agg1);
  {
    long n = (long)N_NODES * HC1;
    k_finalize1<<<(unsigned)((n + 255) / 256), 256, 0, stream>>>(agg1, d1, b1, relu1);
  }

  // ---- GAT layer 2 ----
  k_gemm_bf16<16><<<dim3(N_NODES / 16, HID / 128), 256, 0, stream>>>(relu1, w2t, h2b, HID);
  k_attn_scores<1><<<(N_NODES + 7) / 8, 256, 0, stream>>>(h2b, att_src2, att_dst2, a2s, a2d);
  k_edge_max<1><<<(E_TOT + 255) / 256, 256, 0, stream>>>(ei, a2s, a2d, m2);
  k_edge_msg<1><<<(E_TOT + 7) / 8, 256, 0, stream>>>(ei, a2s, a2d, m2, d2, h2b, agg2);
  {
    long n = (long)N_NODES * HID;
    k_finalize2_pool<<<(unsigned)((n + 255) / 256), 256, 0, stream>>>(agg2, d2, b2, batch, pool);
  }
  k_counts<<<(N_NODES + 255) / 256, 256, 0, stream>>>(batch, cnt);

  // ---- GRU (2 layers, h0=0) + FC ----
  k_gru_fc<<<N_GRAPHS, HID, 0, stream>>>(pool, cnt, Wih0, bih0, bhh0,
                                         Wih1, bih1, bhh1, Wfc, bfc, out);
}
